// GTOutputWithPooling2DoubleAttBig_71330816852710
// MI455X (gfx1250) — matrix operand order repaired and checked by reading, not compiled, not run on hardware
//
#include <hip/hip_runtime.h>

// ---- problem constants (match reference) ----
#define HH   100
#define DD   200          // 2H
#define BB   16
#define LL   2048
#define GG   12
#define QQ   64
#define A8H  800          // 8H
#define DPAD 224          // D padded to 7*32 for K tiles
#define NTILE 13          // ceil(200/16) N tiles over d

typedef __attribute__((ext_vector_type(16))) _Float16 v16h;
typedef __attribute__((ext_vector_type(8)))  float    v8f;

#define WMMA_F16(A, Bm, C) \
    __builtin_amdgcn_wmma_f32_16x16x32_f16(false, (A), false, (Bm), (short)0, (C), false, false)

// ------------------------------------------------------------------
// Kernel 1: global pools over L (pool0_att 16H, pool0_mod 4H).
// Pure bandwidth: one pass over att (104MB) + mod (26MB).
// ------------------------------------------------------------------
__global__ __launch_bounds__(256)
void pool0_kernel(const float* __restrict__ att, const float* __restrict__ mod,
                  const int* __restrict__ mask,
                  float* __restrict__ pool0_att, float* __restrict__ pool0_mod) {
    int b = blockIdx.x, chunk = blockIdx.y, tid = threadIdx.x;
    __shared__ int red[256];
    __shared__ float s_cnt;
    int m = 0;
    for (int l = tid; l < LL; l += 256) m += mask[b * LL + l];
    red[tid] = m; __syncthreads();
    for (int s = 128; s > 0; s >>= 1) { if (tid < s) red[tid] += red[tid + s]; __syncthreads(); }
    if (tid == 0) s_cnt = (float)red[0];
    __syncthreads();
    float cnt = s_cnt;

    if (chunk < 4) {
        int d = chunk * 256 + tid;
        if (d >= A8H) return;
        const float* p = att + (size_t)b * LL * A8H + d;
        float mx = -INFINITY, sm = 0.f;
        for (int l = 0; l < LL; ++l) {
            __builtin_prefetch(p + (size_t)(l + 16) * A8H, 0, 0);
            float v = p[(size_t)l * A8H];
            mx = fmaxf(mx, v); sm += v;
        }
        pool0_att[b * 1600 + d]       = mx;
        pool0_att[b * 1600 + 800 + d] = sm / cnt;
    } else {
        int d = tid;
        if (d >= DD) return;
        const float* p = mod + (size_t)b * LL * DD + d;
        float mx = -INFINITY, sm = 0.f;
        for (int l = 0; l < LL; ++l) {
            float v = p[(size_t)l * DD];
            mx = fmaxf(mx, v); sm += v;
        }
        pool0_mod[b * 400 + d]       = mx;
        pool0_mod[b * 400 + 200 + d] = sm / cnt;
    }
}

// ------------------------------------------------------------------
// Kernel 2: one wave per (batch, y). y in [0,10]: windowed BiDAF step
// (gid = gap[b][1+y]) with fused max/mean pooling and closed-form
// out-of-window correction. y == 11: "gaps" BiDAF on the 12 gap rows,
// writing att2_gaps. All GEMMs via v_wmma_f32_16x16x32_f16.
// ------------------------------------------------------------------
__global__ __launch_bounds__(32)
void bidaf_kernel(const float* __restrict__ mod, const float* __restrict__ att,
                  const int* __restrict__ gap, const int* __restrict__ mask,
                  const float* __restrict__ q_enc, const int* __restrict__ q_mask,
                  const float* __restrict__ cw, const float* __restrict__ qw,
                  const float* __restrict__ cqw, const float* __restrict__ bias_p,
                  float* __restrict__ pa, float* __restrict__ pm,
                  float* __restrict__ p2, float* __restrict__ att2_gaps) {
    __shared__ _Float16 qh[QQ * DPAD];    // q_enc (f16, padded)      28672 B
    __shared__ _Float16 ch[32 * DPAD];    // window rows of mod (f16) 14336 B
    __shared__ _Float16 qct[QQ * 16];     // streamed QC N-tile        2048 B
    __shared__ float    sS[32 * QQ];      // S, later P_q              8192 B
    __shared__ float    sPC[32 * QQ];     // P_c                       8192 B
    __shared__ float    sS1[QQ];
    __shared__ float    sAout[DPAD];
    __shared__ float    sCqw[DPAD];
    __shared__ float    sS0[32];
    __shared__ int      sRows[32];
    __shared__ int      sNW;

    const int b = blockIdx.x, y = blockIdx.y;
    const bool gaps = (y == 11);
    const int t = y;
    const int lane = threadIdx.x;
    const int lh = lane >> 4;         // half of wave
    const int r  = lane & 15;
    const float bias = bias_p[0];

    for (int d = lane; d < DPAD; d += 32) sCqw[d] = (d < DD) ? cqw[d] : 0.f;

    // q_enc -> f16 LDS (zero padded)
    for (int i = lane; i < QQ * DPAD; i += 32) {
        int q = i / DPAD, d = i % DPAD;
        qh[i] = (d < DD) ? (_Float16)q_enc[((size_t)b * QQ + q) * DD + d] : (_Float16)0.f;
    }
    __syncthreads();

    // s1[q] = q . qw + bias
    for (int q = lane; q < QQ; q += 32) {
        float s = 0.f;
        for (int d = 0; d < DD; ++d) s += (float)qh[q * DPAD + d] * qw[d];
        sS1[q] = s + bias;
    }
    __syncthreads();

    // p_out / a_out (attended vector for any all-zero c-row)
    float mx1 = -INFINITY;
    for (int q = 0; q < QQ; ++q) if (q_mask[b * QQ + q] > 0) mx1 = fmaxf(mx1, sS1[q]);
    float sum1 = 0.f;
    for (int q = 0; q < QQ; ++q) if (q_mask[b * QQ + q] > 0) sum1 += __expf(sS1[q] - mx1);
    for (int d = lane; d < DPAD; d += 32) {
        float a = 0.f;
        for (int q = 0; q < QQ; ++q)
            if (q_mask[b * QQ + q] > 0)
                a += (__expf(sS1[q] - mx1) / sum1) * (float)qh[q * DPAD + d];
        sAout[d] = a;
    }

    // build row list
    if (lane == 0) {
        int n = 0;
        if (gaps) {
            for (int j = 0; j < GG; ++j) sRows[n++] = gap[b * GG + j];
        } else {
            int gid = gap[b * GG + 1 + t];
            int lo = gid - 12; if (lo < 0) lo = 0;
            int hi = gid + 12; if (hi > LL - 1) hi = LL - 1;
            for (int l = lo; l <= hi; ++l) if (mask[b * LL + l] > 0) sRows[n++] = l;
        }
        sNW = n;
    }
    __syncthreads();
    const int nW = sNW;

    // c rows -> f16 LDS (zero padded rows/cols)
    for (int i = lane; i < 32 * DPAD; i += 32) {
        int row = i / DPAD, d = i % DPAD;
        _Float16 v = (_Float16)0.f;
        if (row < nW && d < DD) v = (_Float16)mod[((size_t)b * LL + sRows[row]) * DD + d];
        ch[i] = v;
    }
    __syncthreads();

    // s0[l] = c . cw
    {
        float s = 0.f;
        for (int d = 0; d < DD; ++d) s += (float)ch[lane * DPAD + d] * cw[d];
        sS0[lane] = s;
    }
    __syncthreads();

    // ---- GEMM1: S = (C .* cqw) @ Q^T  (32x64), K = 224 ----
    for (int mt = 0; mt < 2; ++mt) {
        for (int nq = 0; nq < 4; ++nq) {
            v8f acc = {};
            for (int kt = 0; kt < 7; ++kt) {
                v16h af, bf;
                int arow = mt * 16 + r;
                #pragma unroll
                for (int e = 0; e < 8; ++e) {
                    int k0 = kt * 32 + lh * 16 + e;
                    int k1 = kt * 32 + lh * 16 + 8 + e;
                    af[e]     = (_Float16)((float)ch[arow * DPAD + k0] * sCqw[k0]);
                    af[e + 8] = (_Float16)((float)ch[arow * DPAD + k1] * sCqw[k1]);
                }
                int bq = nq * 16 + r;
                #pragma unroll
                for (int e = 0; e < 16; ++e) bf[e] = qh[bq * DPAD + kt * 32 + lh * 16 + e];
                acc = WMMA_F16(af, bf, acc);
            }
            #pragma unroll
            for (int v = 0; v < 8; ++v) {
                int row = mt * 16 + lh * 8 + v, col = nq * 16 + r;
                sS[row * QQ + col] = acc[v] + sS0[row] + sS1[col];  // sS1 holds bias
            }
        }
    }
    __syncthreads();

    // ---- P_c: softmax over window rows, per column q ----
    for (int qi = lane; qi < QQ; qi += 32) {
        float mx = -INFINITY;
        for (int l = 0; l < nW; ++l) mx = fmaxf(mx, sS[l * QQ + qi]);
        float sm = 0.f;
        for (int l = 0; l < nW; ++l) sm += __expf(sS[l * QQ + qi] - mx);
        for (int l = 0; l < 32; ++l)
            sPC[l * QQ + qi] = (l < nW) ? __expf(sS[l * QQ + qi] - mx) / sm : 0.f;
    }
    __syncthreads();

    // ---- P_q: softmax over q (masked), in place over sS ----
    {
        int l = lane;
        float mx = -INFINITY;
        for (int q = 0; q < QQ; ++q) if (q_mask[b * QQ + q] > 0) mx = fmaxf(mx, sS[l * QQ + q]);
        float sm = 0.f;
        for (int q = 0; q < QQ; ++q) if (q_mask[b * QQ + q] > 0) sm += __expf(sS[l * QQ + q] - mx);
        for (int q = 0; q < QQ; ++q) {
            float p = (q_mask[b * QQ + q] > 0) ? __expf(sS[l * QQ + q] - mx) / sm : 0.f;
            sS[l * QQ + q] = p;
        }
    }
    __syncthreads();

    // ---- resident A-fragments: P_q (2x2 tiles), P_c^T (4 tiles) ----
    v16h pqf[2][2];
    for (int mt = 0; mt < 2; ++mt)
        for (int kt = 0; kt < 2; ++kt) {
            v16h f;
            int arow = mt * 16 + r;
            #pragma unroll
            for (int e = 0; e < 8; ++e) {
                f[e]     = (_Float16)sS[arow * QQ + kt * 32 + lh * 16 + e];
                f[e + 8] = (_Float16)sS[arow * QQ + kt * 32 + lh * 16 + 8 + e];
            }
            pqf[mt][kt] = f;
        }
    v16h pcf[4];
    for (int mt = 0; mt < 4; ++mt) {
        v16h f;
        int aq = mt * 16 + r;
        #pragma unroll
        for (int e = 0; e < 8; ++e) {
            f[e]     = (_Float16)sPC[(lh * 16 + e) * QQ + aq];
            f[e + 8] = (_Float16)sPC[(lh * 16 + 8 + e) * QQ + aq];
        }
        pcf[mt] = f;
    }

    const float cnt    = (float)nW;
    const float outCnt = (float)(LL - nW);

    // ---- per-16-col tile: A = Pq@Q, QC = Pc^T@C, B2 = Pq@QC, fused pool ----
    for (int no = 0; no < NTILE; ++no) {
        v8f afr[2];
        for (int mt = 0; mt < 2; ++mt) {
            v8f acc = {};
            for (int kt = 0; kt < 2; ++kt) {
                v16h bf;
                #pragma unroll
                for (int e = 0; e < 16; ++e)
                    bf[e] = qh[(kt * 32 + lh * 16 + e) * DPAD + no * 16 + r];
                acc = WMMA_F16(pqf[mt][kt], bf, acc);
            }
            afr[mt] = acc;
        }
        for (int mt = 0; mt < 4; ++mt) {
            v16h bf;
            #pragma unroll
            for (int e = 0; e < 16; ++e)
                bf[e] = ch[(lh * 16 + e) * DPAD + no * 16 + r];
            v8f acc = {};
            acc = WMMA_F16(pcf[mt], bf, acc);
            #pragma unroll
            for (int v = 0; v < 8; ++v)
                qct[(mt * 16 + lh * 8 + v) * 16 + r] = (_Float16)acc[v];
        }
        __syncthreads();
        v8f bfr[2];
        for (int mt = 0; mt < 2; ++mt) {
            v8f acc = {};
            for (int kt = 0; kt < 2; ++kt) {
                v16h bf;
                #pragma unroll
                for (int e = 0; e < 16; ++e)
                    bf[e] = qct[(kt * 32 + lh * 16 + e) * 16 + r];
                acc = WMMA_F16(pqf[mt][kt], bf, acc);
            }
            bfr[mt] = acc;
        }
        __syncthreads();

        int d = no * 16 + r;
        if (gaps) {
            for (int mt = 0; mt < 2; ++mt)
                #pragma unroll
                for (int v = 0; v < 8; ++v) {
                    int row = mt * 16 + lh * 8 + v;
                    if (row < nW && d < DD) {
                        float cv = (float)ch[row * DPAD + d];
                        float av = afr[mt][v], bv = bfr[mt][v];
                        float* o = att2_gaps + ((size_t)b * GG + row) * A8H;
                        o[d] = cv; o[200 + d] = av; o[400 + d] = cv * av; o[600 + d] = cv * bv;
                    }
                }
        } else {
            float mxc = -INFINITY, smc = 0.f, mxa = -INFINITY, sma = 0.f;
            float mxca = -INFINITY, smca = 0.f, mxcb = -INFINITY, smcb = 0.f;
            for (int mt = 0; mt < 2; ++mt)
                #pragma unroll
                for (int v = 0; v < 8; ++v) {
                    int row = mt * 16 + lh * 8 + v;
                    if (row < nW) {
                        float cv = (float)ch[row * DPAD + d];
                        float av = afr[mt][v], bv = bfr[mt][v];
                        mxc = fmaxf(mxc, cv); smc += cv;
                        mxa = fmaxf(mxa, av); sma += av;
                        float ca = cv * av; mxca = fmaxf(mxca, ca); smca += ca;
                        float cb = cv * bv; mxcb = fmaxf(mxcb, cb); smcb += cb;
                    }
                }
            mxc  = fmaxf(mxc,  __shfl_xor(mxc,  16, 32)); smc  += __shfl_xor(smc,  16, 32);
            mxa  = fmaxf(mxa,  __shfl_xor(mxa,  16, 32)); sma  += __shfl_xor(sma,  16, 32);
            mxca = fmaxf(mxca, __shfl_xor(mxca, 16, 32)); smca += __shfl_xor(smca, 16, 32);
            mxcb = fmaxf(mxcb, __shfl_xor(mxcb, 16, 32)); smcb += __shfl_xor(smcb, 16, 32);
            if (lh == 0 && d < DD) {
                float aout = sAout[d];
                float* ppm = pm + ((size_t)b * 11 + t) * 400;
                ppm[d] = fmaxf(0.f, mxc); ppm[200 + d] = smc / cnt;
                float* pp2 = p2 + ((size_t)b * 11 + t) * 1600;
                pp2[d]       = fmaxf(0.f, mxc);   pp2[800 + d]       = smc / cnt;
                pp2[200 + d] = fmaxf(mxa, aout);  pp2[800 + 200 + d] = (sma + outCnt * aout) / cnt;
                pp2[400 + d] = fmaxf(0.f, mxca);  pp2[800 + 400 + d] = smca / cnt;
                pp2[600 + d] = fmaxf(0.f, mxcb);  pp2[800 + 600 + d] = smcb / cnt;
            }
        }
    }

    // ---- windowed att pool (pa): only window rows are non-zero ----
    if (!gaps) {
        float* ppa = pa + ((size_t)b * 11 + t) * 1600;
        for (int d = lane; d < A8H; d += 32) {
            float mx = -INFINITY, sm = 0.f;
            for (int i = 0; i < nW; ++i) {
                float v = att[((size_t)b * LL + sRows[i]) * A8H + d];
                mx = fmaxf(mx, v); sm += v;
            }
            ppa[d] = fmaxf(0.f, mx); ppa[800 + d] = sm / cnt;
        }
    }
}

// ------------------------------------------------------------------
// Kernel 3: final logits — dot products of the concatenated features.
// ------------------------------------------------------------------
__global__ __launch_bounds__(256)
void logits_kernel(const float* __restrict__ att, const float* __restrict__ mod,
                   const int* __restrict__ gap,
                   const float* __restrict__ att2_gaps,
                   const float* __restrict__ pool0_att, const float* __restrict__ pool0_mod,
                   const float* __restrict__ pa, const float* __restrict__ pm,
                   const float* __restrict__ p2,
                   const float* w_att, const float* b_att,
                   const float* w_mod, const float* b_mod,
                   const float* w_att2, const float* b_att2,
                   const float* w_att_s, const float* b_att_s,
                   const float* w_mod_s, const float* b_mod_s,
                   const float* w_att_s2, const float* b_att_s2,
                   float* __restrict__ out) {
    int b = blockIdx.x, j = blockIdx.y, tid = threadIdx.x;
    int gid = gap[b * GG + j];
    const float* wa  = (j == 0) ? w_att_s  : w_att;
    const float* wm_ = (j == 0) ? w_mod_s  : w_mod;
    const float* wa2 = (j == 0) ? w_att_s2 : w_att2;

    float acc = 0.f;
    const float* ag = att + ((size_t)b * LL + gid) * A8H;
    for (int d = tid; d < 800; d += 256) acc += ag[d] * wa[d];
    const float* pav = (j == 0) ? (pool0_att + b * 1600) : (pa + ((size_t)b * 11 + (j - 1)) * 1600);
    for (int d = tid; d < 1600; d += 256) acc += pav[d] * wa[800 + d];
    const float* mg = mod + ((size_t)b * LL + gid) * DD;
    for (int d = tid; d < 200; d += 256) acc += mg[d] * wm_[d];
    const float* pmv = (j == 0) ? (pool0_mod + b * 400) : (pm + ((size_t)b * 11 + (j - 1)) * 400);
    for (int d = tid; d < 400; d += 256) acc += pmv[d] * wm_[200 + d];
    const float* a2g = att2_gaps + ((size_t)b * GG + j) * A8H;
    for (int d = tid; d < 800; d += 256) acc += a2g[d] * wa2[d];
    if (j > 0) {
        const float* p2v = p2 + ((size_t)b * 11 + (j - 1)) * 1600;
        for (int d = tid; d < 1600; d += 256) acc += p2v[d] * wa2[800 + d];
    }

    __shared__ float red[256];
    red[tid] = acc; __syncthreads();
    for (int s = 128; s > 0; s >>= 1) { if (tid < s) red[tid] += red[tid + s]; __syncthreads(); }
    if (tid == 0) {
        float bsum = (j == 0) ? (b_att_s[0] + b_mod_s[0] + b_att_s2[0])
                              : (b_att[0]   + b_mod[0]   + b_att2[0]);
        out[b * GG + j] = red[0] + bsum;
    }
}

// ------------------------------------------------------------------
extern "C" void kernel_launch(void* const* d_in, const int* in_sizes, int n_in,
                              void* d_out, int out_size, void* d_ws, size_t ws_size,
                              hipStream_t stream) {
    const float* att      = (const float*)d_in[0];
    const float* mod      = (const float*)d_in[1];
    const int*   gap      = (const int*)  d_in[2];
    const int*   mask     = (const int*)  d_in[3];
    const float* q_enc    = (const float*)d_in[4];
    const int*   q_mask   = (const int*)  d_in[5];
    const float* c_w      = (const float*)d_in[6];
    const float* q_w      = (const float*)d_in[7];
    const float* cq_w     = (const float*)d_in[8];
    const float* bd_bias  = (const float*)d_in[9];
    const float* w_att    = (const float*)d_in[10];
    const float* b_att    = (const float*)d_in[11];
    const float* w_mod    = (const float*)d_in[12];
    const float* b_mod    = (const float*)d_in[13];
    const float* w_att2   = (const float*)d_in[14];
    const float* b_att2   = (const float*)d_in[15];
    const float* w_att_s  = (const float*)d_in[16];
    const float* b_att_s  = (const float*)d_in[17];
    const float* w_mod_s  = (const float*)d_in[18];
    const float* b_mod_s  = (const float*)d_in[19];
    const float* w_att_s2 = (const float*)d_in[20];
    const float* b_att_s2 = (const float*)d_in[21];
    float* out = (float*)d_out;

    float* ws        = (float*)d_ws;
    float* pool0_att = ws;              // B*1600 = 25600
    float* pool0_mod = ws + 25600;      // B*400  =  6400
    float* pa        = ws + 32000;      // B*11*1600 = 281600
    float* pm        = ws + 313600;     // B*11*400  =  70400
    float* p2        = ws + 384000;     // B*11*1600 = 281600
    float* att2g     = ws + 665600;     // B*12*800  = 153600

    pool0_kernel<<<dim3(BB, 5), 256, 0, stream>>>(att, mod, mask, pool0_att, pool0_mod);
    bidaf_kernel<<<dim3(BB, 12), 32, 0, stream>>>(mod, att, gap, mask, q_enc, q_mask,
                                                  c_w, q_w, cq_w, bd_bias,
                                                  pa, pm, p2, att2g);
    logits_kernel<<<dim3(BB, GG), 256, 0, stream>>>(att, mod, gap, att2g,
                                                    pool0_att, pool0_mod, pa, pm, p2,
                                                    w_att, b_att, w_mod, b_mod, w_att2, b_att2,
                                                    w_att_s, b_att_s, w_mod_s, b_mod_s,
                                                    w_att_s2, b_att_s2, out);
}
